// GNN4KT_41704132444978
// MI455X (gfx1250) — matrix operand order repaired and checked by reading, not compiled, not run on hardware
//
#include <hip/hip_runtime.h>
#include <hip/hip_bf16.h>
#include <stdint.h>

// ---------------------------------------------------------------------------
// GNN4KT on MI455X (gfx1250): all GEMMs via v_wmma_f32_16x16x32_bf16,
// LDS staging via global_load_async_to_lds_b128 (ASYNCcnt), wave32 layout.
// ---------------------------------------------------------------------------

#define BDIM 256
#define BM 128
#define BN 128
#define BK 32
#define PITCH 40          // bf16 elements per LDS row (32 data + 8 pad)

#define NB   256          // batch
#define SS   199          // S
#define SEQ  200          // S+1
#define EH   512          // E == H
#define G4   2048         // 4*H
#define NL   5
#define MROWS (NB*SS)     // 50944
#define SIGMA_C 0.5f

typedef __bf16 bf16;
typedef __bf16 v16bf __attribute__((ext_vector_type(16)));
typedef float  v8f   __attribute__((ext_vector_type(8)));

struct __align__(16) U16 { unsigned int x, y, z, w; };

// ---- CDNA5 async global->LDS (GV mode: 64-bit vaddr, saddr=off) -----------
__device__ __forceinline__ void async_ld_b128(uint32_t lds_off, const void* g) {
  asm volatile("global_load_async_to_lds_b128 %0, %1, off"
               :: "v"(lds_off), "v"((uint64_t)(uintptr_t)g)
               : "memory");
}
#define WAIT_ASYNC(n) asm volatile("s_wait_asynccnt %0" :: "i"(n) : "memory")

__device__ __forceinline__ float sigf(float x) { return 1.f / (1.f + __expf(-x)); }

// ---------------------------------------------------------------------------
// Tile stage: copy BM rows x BK cols of bf16 from global (row-major, ld) into
// LDS (PITCH-padded). Rows clamped (results discarded by store guard); K tail
// clamped to stay in-range (fixed up with zeros post-wait). 2 asyncs/thread.
// ---------------------------------------------------------------------------
__device__ __forceinline__ void stage_tile(bf16* lds, const bf16* g, int ld,
                                           int row0, int nrows, int kbase, int K,
                                           int tid) {
#pragma unroll
  for (int i = 0; i < 2; ++i) {
    int chunk = tid + i * BDIM;       // 0..511
    int r = chunk >> 2;
    int c = (chunk & 3) * 8;          // bf16 column within tile
    int grow = row0 + r; if (grow >= nrows) grow = nrows - 1;
    int k = kbase + c;
    int kcl = K - 8;                  // all our K >= 200
    if (k > kcl) k = kcl & ~7;
    async_ld_b128((uint32_t)(uintptr_t)(lds + r * PITCH + c),
                  g + (long long)grow * ld + k);
  }
}

struct GemmP {
  const bf16* A; const bf16* Bt;      // A [M,K] lda ; Bt [N,K] ldb (B transposed)
  float* C; bf16* Obf; bf16* ObfT;    // optional outputs
  const float* bias;
  long long sA, sB, sC, sO, sOT;      // batch strides (elements)
  int M, N, K, lda, ldb, ldc;
  int act;                            // 0 none, 1 relu
};

__global__ __launch_bounds__(BDIM) void gemm_bf16(GemmP p) {
  __shared__ bf16 tileA[2][BM * PITCH];
  __shared__ bf16 tileB[2][BN * PITCH];
  const int tid  = threadIdx.x;
  const int bz   = blockIdx.z;
  const bf16* Ab = p.A  + (long long)bz * p.sA;
  const bf16* Bb = p.Bt + (long long)bz * p.sB;
  const int m0 = blockIdx.x * BM;
  const int n0 = blockIdx.y * BN;
  const int lane = tid & 31;
  const int wave = tid >> 5;
  const int wm = (wave & 1) * 64;     // 2 waves along M, 4 along N
  const int wn = (wave >> 1) * 32;

  v8f acc[4][2] = {};
  const int nkb = (p.K + BK - 1) / BK;

  stage_tile(tileA[0], Ab, p.lda, m0, p.M, 0, p.K, tid);
  stage_tile(tileB[0], Bb, p.ldb, n0, p.N, 0, p.K, tid);

  for (int kb = 0; kb < nkb; ++kb) {
    const int s = kb & 1;
    const bool pref = (kb + 1 < nkb);
    if (pref) {
      stage_tile(tileA[s ^ 1], Ab, p.lda, m0, p.M, (kb + 1) * BK, p.K, tid);
      stage_tile(tileB[s ^ 1], Bb, p.ldb, n0, p.N, (kb + 1) * BK, p.K, tid);
      WAIT_ASYNC(4);                  // drain current stage, keep prefetch inflight
    } else {
      WAIT_ASYNC(0);
    }
    __syncthreads();

    const int kbase = kb * BK;
    if (kbase + BK > p.K) {           // zero K tail (adj GEMM, K=200)
      int kv = p.K - kbase, w = BK - kv;
      for (int idx = tid; idx < BM * w; idx += BDIM) {
        int r = idx / w, c = kv + idx % w;
        tileA[s][r * PITCH + c] = (bf16)0.f;
        tileB[s][r * PITCH + c] = (bf16)0.f;
      }
      __syncthreads();
    }

    // Fragments per ISA 16-bit layouts:
    //  A 16x32: lane(0-15) row=lane, K kh..kh+7 & 16+kh..; kh = (lane>>4)*8
    //  B 32x16 (from [N,K] LDS): lane col=lane&15, K kh..kh+7 & 16+kh.. (same as A)
    union AF { v16bf v; U16 q[2]; };
    AF a[4], b[2];
    const int rsel = lane & 15;
    const int kh   = (lane >> 4) * 8;
    const int kb2  = (lane >> 4) * 8;
#pragma unroll
    for (int tm = 0; tm < 4; ++tm) {
      const bf16* pa = &tileA[s][(wm + tm * 16 + rsel) * PITCH + kh];
      a[tm].q[0] = *(const U16*)pa;
      a[tm].q[1] = *(const U16*)(pa + 16);
    }
#pragma unroll
    for (int tn = 0; tn < 2; ++tn) {
      const bf16* pb = &tileB[s][(wn + tn * 16 + rsel) * PITCH + kb2];
      b[tn].q[0] = *(const U16*)pb;
      b[tn].q[1] = *(const U16*)(pb + 16);
    }
#pragma unroll
    for (int tm = 0; tm < 4; ++tm)
#pragma unroll
      for (int tn = 0; tn < 2; ++tn)
        acc[tm][tn] = __builtin_amdgcn_wmma_f32_16x16x32_bf16(
            false, a[tm].v, false, b[tn].v, (short)0, acc[tm][tn], false, false);
    __syncthreads();
  }

  // Epilogue: C/D layout — VGPR r, lane<16: M=r, lane>=16: M=8+r; N=lane&15
#pragma unroll
  for (int tm = 0; tm < 4; ++tm)
#pragma unroll
    for (int tn = 0; tn < 2; ++tn) {
      int col  = n0 + wn + tn * 16 + (lane & 15);
      int rowb = m0 + wm + tm * 16 + (lane >> 4) * 8;
      if (col < p.N) {
        float bv = p.bias ? p.bias[col] : 0.f;
#pragma unroll
        for (int r = 0; r < 8; ++r) {
          int row = rowb + r;
          if (row < p.M) {
            float v = acc[tm][tn][r] + bv;
            if (p.act == 1) v = fmaxf(v, 0.f);
            if (p.C)    p.C[(long long)bz * p.sC + (long long)row * p.ldc + col] = v;
            if (p.Obf)  p.Obf[(long long)bz * p.sO + (long long)row * p.N + col] = (bf16)v;
            if (p.ObfT) p.ObfT[(long long)bz * p.sOT + (long long)col * p.M + row] = (bf16)v;
          }
        }
      }
    }
}

// ---------------------------------------------------------------------------
// Elementwise / conversion kernels
// ---------------------------------------------------------------------------
__global__ void embed_k(const int* q, const int* c, const int* r,
                        const int* qs, const int* cs, const int* rs,
                        const float* Wq, const float* Wc, const float* Wr,
                        bf16* qa, bf16* qemb) {
  long long i = (long long)blockIdx.x * BDIM + threadIdx.x;  // NB*SEQ*EH
  int h = i & 511;
  long long t2 = i >> 9; int t = (int)(t2 % SEQ); int b = (int)(t2 / SEQ);
  int pid = (t == 0) ? q[b * SS] : qs[b * SS + t - 1];
  int cid = (t == 0) ? c[b * SS] : cs[b * SS + t - 1];
  int tg  = (t == 0) ? r[b * SS] : rs[b * SS + t - 1];
  float qe  = Wq[(long long)pid * EH + h] + Wc[(long long)cid * EH + h];
  float qav = qe + Wr[tg * EH + h];
  qemb[((long long)b * SEQ + t) * EH + h] = (bf16)qe;     // [b][t][h]
  qa  [((long long)t * NB  + b) * EH + h] = (bf16)qav;    // [t][b][h]
}

__global__ void build_wcomb(const float* Wih, const float* Whh,
                            const float* bih, const float* bhh,
                            bf16* wcomb, float* bsum) {
  long long i = (long long)blockIdx.x * BDIM + threadIdx.x;  // 5*2048*1024
  int k = (int)(i & 1023); long long t = i >> 10;
  int n = (int)(t & 2047); int l = (int)(t >> 11);
  float v = (k < 512) ? Whh[((long long)l * G4 + n) * EH + k]
                      : Wih[((long long)l * G4 + n) * EH + (k - 512)];
  wcomb[i] = (bf16)v;
  if (i < NL * G4) bsum[i] = bih[i] + bhh[i];
}

__global__ void conv_t(const float* src, bf16* dst, int K, int N) {
  long long i = (long long)blockIdx.x * BDIM + threadIdx.x;  // over N*K
  if (i >= (long long)K * N) return;
  int k = (int)(i % K); long long n = i / K;
  dst[i] = (bf16)src[(long long)k * N + n];
}

__global__ void conv_d(const float* src, bf16* dst, long long n) {
  long long i = (long long)blockIdx.x * BDIM + threadIdx.x;
  if (i < n) dst[i] = (bf16)src[i];
}

__global__ void zero_init(float* cst, bf16* hcat) {
  int i = blockIdx.x * BDIM + threadIdx.x;   // 131072
  cst[i] = 0.f;
  int b = i >> 9, h = i & 511;
  hcat[b * 1024 + h] = (bf16)0.f;
}

__global__ void copy_xt(bf16* hcat, const bf16* xt) {
  int i = blockIdx.x * BDIM + threadIdx.x;   // 131072
  int b = i >> 9, h = i & 511;
  hcat[b * 1024 + 512 + h] = xt[i];
}

__global__ void lstm_cell(const float* gates, float* cst, bf16* hcat,
                          bf16* xout, float* outf, bf16* outb, int t) {
  int i = blockIdx.x * BDIM + threadIdx.x;   // 131072
  int b = i >> 9, h = i & 511;
  const float* g = gates + b * G4;
  float ig = sigf(g[h]);
  float fg = sigf(g[512 + h]);
  float gg = tanhf(g[1024 + h]);
  float og = sigf(g[1536 + h]);
  float cv = fg * cst[i] + ig * gg;
  float hn = og * tanhf(cv);
  cst[i] = cv;
  hcat[b * 1024 + h] = (bf16)hn;
  if (xout) xout[i] = (bf16)hn;                                 // [t][b][h]
  if (outf) {
    long long o = ((long long)b * SEQ + t) * EH + h;            // [b][t][h]
    outf[o] = hn; outb[o] = (bf16)hn;
  }
}

__global__ void mix_k(const float* h, const float* out, bf16* mixb) {
  long long i = (long long)blockIdx.x * BDIM + threadIdx.x;     // NB*SEQ*EH
  mixb[i] = (bf16)((1.f - SIGMA_C) * h[i] + SIGMA_C * out[i]);
}

__global__ void concat_k(const float* hf, const bf16* qemb, bf16* xm) {
  long long i = (long long)blockIdx.x * BDIM + threadIdx.x;     // MROWS*1024
  int j = (int)(i & 1023); long long row = i >> 10;
  int s = (int)(row % SS); int b = (int)(row / SS);
  long long src = ((long long)b * SEQ + (s + 1)) * EH;
  xm[i] = (j < 512) ? (bf16)hf[src + j] : qemb[src + (j - 512)];
}

__global__ void final_k(const bf16* h2, const float* W3, const float* b3, float* y) {
  int row = blockIdx.x * BDIM + threadIdx.x;
  if (row >= MROWS) return;
  float s = b3[0];
  const bf16* hp = h2 + (long long)row * 256;
  for (int k = 0; k < 256; ++k) s += (float)hp[k] * W3[k];
  y[row] = sigf(s);
}

// ---------------------------------------------------------------------------
static void launch_gemm(hipStream_t st, dim3 grid,
                        const bf16* A, long long sA, int lda,
                        const bf16* Bt, long long sB, int ldb,
                        float* C, long long sC, int ldc,
                        const float* bias,
                        bf16* obf, long long sO, bf16* obfT, long long sOT,
                        int M, int N, int K, int act) {
  GemmP p{A, Bt, C, obf, obfT, bias, sA, sB, sC, sO, sOT, M, N, K, lda, ldb, ldc, act};
  gemm_bf16<<<grid, BDIM, 0, st>>>(p);
}

extern "C" void kernel_launch(void* const* d_in, const int* in_sizes, int n_in,
                              void* d_out, int out_size, void* d_ws, size_t ws_size,
                              hipStream_t stream) {
  const int*   q    = (const int*)d_in[0];
  const int*   c    = (const int*)d_in[1];
  const int*   r    = (const int*)d_in[2];
  const int*   qs   = (const int*)d_in[3];
  const int*   cs   = (const int*)d_in[4];
  const int*   rs   = (const int*)d_in[5];
  const float* adj  = (const float*)d_in[6];
  const float* Wq   = (const float*)d_in[7];
  const float* Wc   = (const float*)d_in[8];
  const float* Wr   = (const float*)d_in[9];
  const float* Wih  = (const float*)d_in[10];
  const float* Whh  = (const float*)d_in[11];
  const float* bih  = (const float*)d_in[12];
  const float* bhh  = (const float*)d_in[13];
  const float* Wg   = (const float*)d_in[14];
  const float* W1   = (const float*)d_in[15];
  const float* b1   = (const float*)d_in[16];
  const float* W2   = (const float*)d_in[17];
  const float* b2   = (const float*)d_in[18];
  const float* W3   = (const float*)d_in[19];
  const float* b3   = (const float*)d_in[20];
  float* y = (float*)d_out;

  char* w = (char*)d_ws;
  auto alloc = [&](size_t bytes) -> void* {
    void* p = (void*)w; w += (bytes + 255) & ~(size_t)255; return p;
  };
  const long long NE = (long long)NB * SEQ * EH;              // 26,214,400
  bf16*  qa    = (bf16*)alloc(NE * 2);
  bf16*  qemb  = (bf16*)alloc(NE * 2);
  bf16*  wcomb = (bf16*)alloc((long long)NL * G4 * 1024 * 2);
  float* bsum  = (float*)alloc(NL * G4 * 4);
  bf16*  wgt   = (bf16*)alloc((long long)NL * EH * EH * 2);
  bf16*  adjb  = (bf16*)alloc(SEQ * SEQ * 2);
  bf16*  w1t   = (bf16*)alloc(512 * 1024 * 2);
  bf16*  w2t   = (bf16*)alloc(256 * 512 * 2);
  bf16*  xb0   = (bf16*)alloc(NE * 2);
  bf16*  xb1   = (bf16*)alloc(NE * 2);
  bf16*  hcat  = (bf16*)alloc(NB * 1024 * 2);
  float* cst   = (float*)alloc(NB * EH * 4);
  float* gates = (float*)alloc(NB * G4 * 4);
  float* outf  = (float*)alloc(NE * 4);
  bf16*  outb  = (bf16*)alloc(NE * 2);
  float* hfin  = (float*)alloc(NE * 4);
  bf16*  mixb  = (bf16*)alloc(NE * 2);
  bf16*  t0t   = (bf16*)alloc(NE * 2);
  bf16*  xmlp  = (bf16*)alloc((long long)MROWS * 1024 * 2);
  bf16*  h1b   = (bf16*)alloc((long long)MROWS * 512 * 2);
  bf16*  h2b   = (bf16*)alloc((long long)MROWS * 256 * 2);

  // ---- embeddings + weight conversion -------------------------------------
  embed_k<<<102400, BDIM, 0, stream>>>(q, c, r, qs, cs, rs, Wq, Wc, Wr, qa, qemb);
  build_wcomb<<<40960, BDIM, 0, stream>>>(Wih, Whh, bih, bhh, wcomb, bsum);
  for (int l = 0; l < NL; ++l)
    conv_t<<<1024, BDIM, 0, stream>>>(Wg + (long long)l * EH * EH,
                                      wgt + (long long)l * EH * EH, EH, EH);
  conv_t<<<2048, BDIM, 0, stream>>>(W1, w1t, 1024, 512);
  conv_t<<<512,  BDIM, 0, stream>>>(W2, w2t, 512, 256);
  conv_d<<<157,  BDIM, 0, stream>>>(adj, adjb, (long long)SEQ * SEQ);

  // ---- stacked LSTM: gates = [h|x] @ [Whh|Wih]^T + b ----------------------
  const bf16* Xin = qa;
  bf16* Xout = xb0;
  for (int l = 0; l < NL; ++l) {
    zero_init<<<512, BDIM, 0, stream>>>(cst, hcat);
    const bf16*  wl = wcomb + (long long)l * G4 * 1024;
    const float* bl = bsum + l * G4;
    bool last = (l == NL - 1);
    for (int t = 0; t < SEQ; ++t) {
      copy_xt<<<512, BDIM, 0, stream>>>(hcat, Xin + (long long)t * NB * EH);
      launch_gemm(stream, dim3(2, 16, 1),
                  hcat, 0, 1024, wl, 0, 1024,
                  gates, 0, G4, bl, nullptr, 0, nullptr, 0,
                  NB, G4, 1024, 0);
      lstm_cell<<<512, BDIM, 0, stream>>>(gates, cst, hcat,
                                          last ? nullptr : (Xout + (long long)t * NB * EH),
                                          last ? outf : nullptr,
                                          last ? outb : nullptr, t);
    }
    if (!last) { Xin = Xout; Xout = (Xout == xb0) ? xb1 : xb0; }
  }

  // ---- 5 GCN layers: h = act(adj @ (x @ Wg[i])) ---------------------------
  const long long BS = (long long)SEQ * EH;   // per-batch activation stride
  const bf16* gA = outb;
  for (int i = 0; i < NL; ++i) {
    if (i > 0) {
      mix_k<<<102400, BDIM, 0, stream>>>(hfin, outf, mixb);
      gA = mixb;
    }
    // t0t[b] = (x[b] @ Wg[i])^T  (bf16, [512][200]) for use as B operand
    launch_gemm(stream, dim3(2, 4, NB),
                gA, BS, EH, wgt + (long long)i * EH * EH, 0, EH,
                nullptr, 0, 0, nullptr, nullptr, 0, t0t, BS,
                SEQ, EH, EH, 0);
    // h[b] = act(adj @ t0[b])
    launch_gemm(stream, dim3(2, 4, NB),
                adjb, 0, SEQ, t0t, BS, SEQ,
                hfin, BS, EH, nullptr, nullptr, 0, nullptr, 0,
                SEQ, EH, SEQ, (i < 4) ? 1 : 0);
  }

  // ---- output MLP ---------------------------------------------------------
  concat_k<<<203776, BDIM, 0, stream>>>(hfin, qemb, xmlp);
  launch_gemm(stream, dim3(398, 4, 1),
              xmlp, 0, 1024, w1t, 0, 1024,
              nullptr, 0, 0, b1, h1b, 0, nullptr, 0,
              MROWS, 512, 1024, 1);
  launch_gemm(stream, dim3(398, 2, 1),
              h1b, 0, 512, w2t, 0, 512,
              nullptr, 0, 0, b2, h2b, 0, nullptr, 0,
              MROWS, 256, 512, 1);
  final_k<<<(MROWS + BDIM - 1) / BDIM, BDIM, 0, stream>>>(h2b, W3, b3, y);
}
